// FBNLinear_44813688767076
// MI455X (gfx1250) — compile-verified, matrix operand order checked by reading, not run on hardware
//
#include <hip/hip_runtime.h>

// ---------------------------------------------------------------------------
// y[b,o] = bias[o] + sum_j tanh( x[b,j] * W_eff[o,j] )
// W_eff  = softmax(theta)[1]*W + [2]*tanh(W) + [3]*sin(W)
// B=128, OUT=512, IN=1024, KF=4, all f32.
// ---------------------------------------------------------------------------

typedef __attribute__((ext_vector_type(16))) _Float16 v16h;
typedef __attribute__((ext_vector_type(8)))  float    v8f;

#define B_DIM   128
#define OUT_DIM 512
#define IN_DIM  1024

__device__ __forceinline__ float fast_tanh(float v) {
#if __has_builtin(__builtin_amdgcn_tanhf)
  return __builtin_amdgcn_tanhf(v);
#else
  // tanh(v) = 1 - 2/(exp(2v)+1) ; exp(2v) = exp2(2v*log2e)
  float e = __builtin_amdgcn_exp2f(v * 2.8853900817779268f);
  return 1.0f - 2.0f * __builtin_amdgcn_rcpf(e + 1.0f);
#endif
}

// XOR-swizzle of float index k within a 1024-float LDS row for row m.
// Keeps 4-float groups contiguous (bits [1:0] untouched), permutes bank
// groups (bits [5:2]) by m so 16 lanes reading the same k in 16 different
// rows hit 16 distinct bank groups. LDS stays exactly 16*1024*4 = 64 KB.
__device__ __forceinline__ int swzk(int m, int k) {
  return (k & ~63) | ((k ^ (m << 2)) & 60) | (k & 3);
}

// ---------------------------------------------------------------------------
// Pass 1: W_eff[o,j], one thread per (o,j). 2048 blocks x 256 threads exact.
// ---------------------------------------------------------------------------
__global__ __launch_bounds__(256) void fbn_weff_kernel(
    const float* __restrict__ W, const float* __restrict__ theta,
    float* __restrict__ Weff) {
  const int idx = blockIdx.x * 256 + threadIdx.x;          // < OUT*IN exactly
  const float4 th = ((const float4*)theta)[idx];           // KF=4 contiguous
  const float mx = fmaxf(fmaxf(th.x, th.y), fmaxf(th.z, th.w));
  const float L2E = 1.4426950408889634f;
  const float e0 = __builtin_amdgcn_exp2f((th.x - mx) * L2E); // 'zero' atom
  const float e1 = __builtin_amdgcn_exp2f((th.y - mx) * L2E); // 'id'
  const float e2 = __builtin_amdgcn_exp2f((th.z - mx) * L2E); // 'tanh'
  const float e3 = __builtin_amdgcn_exp2f((th.w - mx) * L2E); // 'sin'
  const float inv = __builtin_amdgcn_rcpf(e0 + e1 + e2 + e3);
  const float w = W[idx];
  const float sw = __builtin_amdgcn_sinf(w * 0.15915494309189535f); // sin(w)
  Weff[idx] = (e1 * w + e2 * fast_tanh(w) + e3 * sw) * inv;
}

// ---------------------------------------------------------------------------
// Pass 2: block = (16 batch rows) x (8 outputs, one per wave32).
// grid = (OUT/8, B/16) = (64, 8), block = 256 threads = 8 waves.
// Reduction over j runs through v_wmma_f32_16x16x32_f16 with B = ones:
// D[m,n] += sum_k t[m,k], accumulated in f32.
// ---------------------------------------------------------------------------
__global__ __launch_bounds__(256) void fbn_main_kernel(
    const float* __restrict__ x, const float* __restrict__ Weff,
    const float* __restrict__ bias, float* __restrict__ y) {
  __shared__ float xs[16 * IN_DIM];                        // 64 KB, swizzled

  const int tid = threadIdx.x;
  const int bt  = blockIdx.y;                              // batch tile

  // ---- cooperative swizzled fill of the 16x1024 x-tile -------------------
  const float4* xg = (const float4*)(x + (size_t)bt * 16 * IN_DIM);
#pragma unroll
  for (int i = 0; i < 16; ++i) {
    const int f    = i * 256 + tid;                        // float4 index
    const int row  = f >> 8;                               // 256 float4 / row
    const int col4 = f & 255;
    const int c4s  = (col4 & ~15) | ((col4 & 15) ^ row);   // == swzk/4
    *(float4*)&xs[row * IN_DIM + c4s * 4] = xg[f];
  }
  __syncthreads();

  const int wave = tid >> 5;
  const int lane = tid & 31;
  const int mrow = lane & 15;          // A-matrix row M this lane feeds
  const int lh   = lane >> 4;          // lane half selects K sub-blocks
  const int koff = lh * 8;             // half 0: K 0..7/16..23; half 1: +8
  const int o    = blockIdx.x * 8 + wave;

  const float* wrow = Weff + (size_t)o * IN_DIM;
  const float* xrow = xs + mrow * IN_DIM;

  v16h ones;
#pragma unroll
  for (int i = 0; i < 16; ++i) ones[i] = (_Float16)1.0f;

  v8f acc0 = {};
  v8f acc1 = {};

  for (int kc = 0; kc < IN_DIM; kc += 64) {                // 2 chunks / iter
#pragma unroll
    for (int c = 0; c < 2; ++c) {
      const int k0 = kc + c * 32 + koff;
      const int k1 = k0 + 16;
      const float4 xa = *(const float4*)&xrow[swzk(mrow, k0)];
      const float4 xb = *(const float4*)&xrow[swzk(mrow, k0 + 4)];
      const float4 xc = *(const float4*)&xrow[swzk(mrow, k1)];
      const float4 xd = *(const float4*)&xrow[swzk(mrow, k1 + 4)];
      const float4 wa = *(const float4*)(wrow + k0);
      const float4 wb = *(const float4*)(wrow + k0 + 4);
      const float4 wc = *(const float4*)(wrow + k1);
      const float4 wd = *(const float4*)(wrow + k1 + 4);

      const float xv[16] = {xa.x, xa.y, xa.z, xa.w, xb.x, xb.y, xb.z, xb.w,
                            xc.x, xc.y, xc.z, xc.w, xd.x, xd.y, xd.z, xd.w};
      const float wv[16] = {wa.x, wa.y, wa.z, wa.w, wb.x, wb.y, wb.z, wb.w,
                            wc.x, wc.y, wc.z, wc.w, wd.x, wd.y, wd.z, wd.w};
      v16h amat;
#pragma unroll
      for (int i = 0; i < 16; ++i)
        amat[i] = (_Float16)fast_tanh(xv[i] * wv[i]);

      if (c == 0)
        acc0 = __builtin_amdgcn_wmma_f32_16x16x32_f16(
            false, amat, false, ones, (short)0, acc0, false, false);
      else
        acc1 = __builtin_amdgcn_wmma_f32_16x16x32_f16(
            false, amat, false, ones, (short)0, acc1, false, false);
    }
  }

  // Every D column holds the identical row-sum; lane with N==0 in each half
  // writes its 8 batch rows (half 0 -> M 0..7, half 1 -> M 8..15).
  if (mrow == 0) {
    const float bo = bias[o];
#pragma unroll
    for (int r = 0; r < 8; ++r) {
      const int brow = bt * 16 + lh * 8 + r;
      y[(size_t)brow * OUT_DIM + o] = acc0[r] + acc1[r] + bo;
    }
  }
}

// ---------------------------------------------------------------------------
extern "C" void kernel_launch(void* const* d_in, const int* in_sizes, int n_in,
                              void* d_out, int out_size, void* d_ws,
                              size_t ws_size, hipStream_t stream) {
  const float* x     = (const float*)d_in[0];  // (128, 1024)
  const float* W     = (const float*)d_in[1];  // (512, 1024)
  const float* b     = (const float*)d_in[2];  // (512,)
  const float* theta = (const float*)d_in[3];  // (512, 1024, 4)
  float* y    = (float*)d_out;                 // (128, 512)
  float* Weff = (float*)d_ws;                  // 512*1024 f32 = 2 MB scratch

  fbn_weff_kernel<<<(OUT_DIM * IN_DIM) / 256, 256, 0, stream>>>(W, theta, Weff);

  dim3 grid(OUT_DIM / 8, B_DIM / 16);
  fbn_main_kernel<<<grid, 256, 0, stream>>>(x, Weff, b, y);
}
